// CliffordHead_22351009809276
// MI455X (gfx1250) — matrix operand order repaired and checked by reading, not compiled, not run on hardware
//
#include <hip/hip_runtime.h>

// ---------------- problem constants (from reference) ----------------
#define BATCH   8192
#define INFEAT  1024
#define HID     2048
#define NTOT    (HID * 4)      // 8192 columns of h
#define BM      128            // batch tile
#define BN      128            // N tile (= 32 multivector channels)
#define KS      32             // K slab per bf16 WMMA step
#define NBLK    (NTOT / BN)    // 64
#define MBLK    (BATCH / BM)   // 64
#define KITERS  (INFEAT / KS)  // 32

#define PLANE_B   10240        // one 128x40 bf16 LDS plane (bytes)
#define BUF_B     (4 * PLANE_B)// one staging buffer: Ahi|Alo|Bhi|Blo = 40 KB
#define SMEM_B    (2 * BUF_B)  // double buffered = 80 KB (>= 66 KB h-tile)

typedef __attribute__((ext_vector_type(16))) __bf16 v16bf;
typedef __attribute__((ext_vector_type(8)))  __bf16 v8bf;
typedef __attribute__((ext_vector_type(4)))  __bf16 v4bf;
typedef __attribute__((ext_vector_type(8)))  float  v8f;

// ---------------- CDNA5 async global->LDS helpers ----------------
// Low 32 bits of a generic pointer into the shared aperture ARE the
// wave-relative LDS byte offset (ISA 10.2).
__device__ __forceinline__ unsigned lds_off(const void* p) {
    return (unsigned)(unsigned long long)p;
}
__device__ __forceinline__ void async_b128(unsigned lds, const void* gaddr) {
    asm volatile("global_load_async_to_lds_b128 %0, %1, off"
                 :: "v"(lds), "v"(gaddr) : "memory");
}
__device__ __forceinline__ void wait_async0() {
#if __has_builtin(__builtin_amdgcn_s_wait_asynccnt)
    __builtin_amdgcn_s_wait_asynccnt(0);
#else
    asm volatile("s_wait_asynccnt 0x0" ::: "memory");
#endif
}

// Assemble a 16-element bf16 WMMA fragment from two 16-byte LDS chunks.
__device__ __forceinline__ v16bf frag16(const __bf16* p0, const __bf16* p1) {
    v8bf a = *(const v8bf*)p0;
    v8bf b = *(const v8bf*)p1;
    return __builtin_shufflevector(a, b, 0,1,2,3,4,5,6,7,8,9,10,11,12,13,14,15);
}

// Split one float4 into bf16 hi/lo and store into LDS staging rows.
__device__ __forceinline__ void split_store(__bf16* hi, __bf16* lo, float4 v) {
    float f[4] = {v.x, v.y, v.z, v.w};
#pragma unroll
    for (int e = 0; e < 4; ++e) {
        __bf16 h = (__bf16)f[e];
        hi[e] = h;
        lo[e] = (__bf16)(f[e] - (float)h);
    }
}

// ---------------- shared epilogue: Clifford quadratic + readout ----------------
// g0 = w0 h0^2 + w3 (h1^2+h2^2) - w7 h3^2 ; g1 = (w1+w4) h0 h1 + (w8-w5) h2 h3
// g2 = (w1+w4) h0 h2 + (w5-w8) h1 h3     ; g3 = (w2+w9) h0 h3  (e1e2 cancels)
__device__ __forceinline__ void clifford_epilogue(
        const float (*hs)[132], float (*red)[2],
        const float* __restrict__ wgp, const float* __restrict__ Wread,
        float* __restrict__ partial, int bN, int bM, int t) {
    const int row   = t >> 1;
    const int cbase = (t & 1) * 16;
    float p0 = 0.f, p1 = 0.f;
#pragma unroll 4
    for (int c = 0; c < 16; ++c) {
        const int cl = cbase + c;
        float4 h = *(const float4*)&hs[row][cl * 4];
        const int n = bN * 32 + cl;
        const float* w = &wgp[n * 10];
        const float w14 = w[1] + w[4];
        const float g0 = w[0]*h.x*h.x + w[3]*(h.y*h.y + h.z*h.z) - w[7]*h.w*h.w;
        const float g1 = w14*h.x*h.y + (w[8] - w[5])*h.z*h.w;
        const float g2 = w14*h.x*h.z + (w[5] - w[8])*h.y*h.w;
        const float g3 = (w[2] + w[9])*h.x*h.w;
        const float* r0 = &Wread[(size_t)0 * NTOT + n * 4];
        const float* r1 = &Wread[(size_t)1 * NTOT + n * 4];
        p0 += g0*r0[0] + g1*r0[1] + g2*r0[2] + g3*r0[3];
        p1 += g0*r1[0] + g1*r1[1] + g2*r1[2] + g3*r1[3];
    }
    red[t][0] = p0;
    red[t][1] = p1;
    __syncthreads();
    if ((t & 1) == 0) {
        const int grow = bM * BM + row;
        const float s0 = red[t][0] + red[t + 1][0];
        const float s1 = red[t][1] + red[t + 1][1];
        const size_t base = (size_t)bN * (BATCH * 2) + (size_t)grow * 2;
        partial[base]     = s0;   // fixed order -> deterministic
        partial[base + 1] = s1;
    }
}

// ---------------- one-time f32 -> bf16 hi/lo split (pre-pass) ----------------
__global__ __launch_bounds__(256)
void bf16_split(const float* __restrict__ src,
                __bf16* __restrict__ hi, __bf16* __restrict__ lo) {
    const size_t i = ((size_t)blockIdx.x * 256 + threadIdx.x) * 4;
    float4 v = *(const float4*)&src[i];
    float f[4] = {v.x, v.y, v.z, v.w};
    v4bf h4, l4;
#pragma unroll
    for (int e = 0; e < 4; ++e) {
        __bf16 h = (__bf16)f[e];
        h4[e] = h;
        l4[e] = (__bf16)(f[e] - (float)h);
    }
    *(v4bf*)&hi[i] = h4;
    *(v4bf*)&lo[i] = l4;
}

// Issue one K-slab of async global->LDS copies (8 x b128 per thread).
__device__ __forceinline__ void stage_slab(
        const __bf16* __restrict__ Ah, const __bf16* __restrict__ Al,
        const __bf16* __restrict__ Bh, const __bf16* __restrict__ Bl,
        unsigned char* buf, int t, int rowA0, int rowB0, int kk) {
    __bf16 (*dAh)[40] = (__bf16 (*)[40])(buf);
    __bf16 (*dAl)[40] = (__bf16 (*)[40])(buf + PLANE_B);
    __bf16 (*dBh)[40] = (__bf16 (*)[40])(buf + 2 * PLANE_B);
    __bf16 (*dBl)[40] = (__bf16 (*)[40])(buf + 3 * PLANE_B);
#pragma unroll
    for (int c = 0; c < 2; ++c) {
        const int e  = c * 256 + t;     // 0..511 chunk id
        const int m  = e >> 2;          // 0..127 tile row
        const int j8 = (e & 3) * 8;     // 0,8,16,24 within K slab
        const size_t ga = (size_t)(rowA0 + m) * INFEAT + kk + j8;
        const size_t gb = (size_t)(rowB0 + m) * INFEAT + kk + j8;
        async_b128(lds_off(&dAh[m][j8]), Ah + ga);
        async_b128(lds_off(&dAl[m][j8]), Al + ga);
        async_b128(lds_off(&dBh[m][j8]), Bh + gb);
        async_b128(lds_off(&dBl[m][j8]), Bl + gb);
    }
}

// ---------------- main fused kernel: async double-buffered WMMA GEMM ----------------
__global__ __launch_bounds__(256)
void clifford_fused_async(const __bf16* __restrict__ Ah, const __bf16* __restrict__ Al,
                          const __bf16* __restrict__ Bh, const __bf16* __restrict__ Bl,
                          const float* __restrict__ bemb, const float* __restrict__ wgp,
                          const float* __restrict__ Wread, float* __restrict__ partial) {
    __shared__ __align__(16) unsigned char smem[SMEM_B];   // 2 staging bufs / h tile
    __shared__ float red[256][2];

    const int t    = threadIdx.x;
    const int bN   = blockIdx.x;
    const int bM   = blockIdx.y;
    const int lane = t & 31;
    const int wid  = t >> 5;
    const int wm   = wid >> 1;          // 0..3
    const int wn   = wid & 1;           // 0..1
    const int l15  = lane & 15;
    const int kh   = lane >> 4;

    v8f acc[2][4];
#pragma unroll
    for (int ti = 0; ti < 2; ++ti)
#pragma unroll
        for (int tj = 0; tj < 4; ++tj) { v8f z = {}; acc[ti][tj] = z; }

    // prologue: prefetch slab 0 into buffer 0
    stage_slab(Ah, Al, Bh, Bl, smem, t, bM * BM, bN * BN, 0);

    for (int it = 0; it < KITERS; ++it) {
        unsigned char* bufc = smem + (it & 1) * BUF_B;
        wait_async0();          // my async loads for current buffer done
        __syncthreads();        // everyone's done; next buffer free for reuse
        if (it + 1 < KITERS)
            stage_slab(Ah, Al, Bh, Bl, smem + ((it + 1) & 1) * BUF_B,
                       t, bM * BM, bN * BN, (it + 1) * KS);

        __bf16 (*cAh)[40] = (__bf16 (*)[40])(bufc);
        __bf16 (*cAl)[40] = (__bf16 (*)[40])(bufc + PLANE_B);
        __bf16 (*cBh)[40] = (__bf16 (*)[40])(bufc + 2 * PLANE_B);
        __bf16 (*cBl)[40] = (__bf16 (*)[40])(bufc + 3 * PLANE_B);

        v16bf ahi[2], alo[2];
#pragma unroll
        for (int ti = 0; ti < 2; ++ti) {
            const int ar = wm * 32 + ti * 16 + l15;
            ahi[ti] = frag16(&cAh[ar][kh * 8], &cAh[ar][16 + kh * 8]);
            alo[ti] = frag16(&cAl[ar][kh * 8], &cAl[ar][16 + kh * 8]);
        }
#pragma unroll
        for (int tj = 0; tj < 4; ++tj) {
            const int br = wn * 64 + tj * 16 + l15;
            v16bf bhi = frag16(&cBh[br][kh * 8], &cBh[br][16 + kh * 8]);
            v16bf blo = frag16(&cBl[br][kh * 8], &cBl[br][16 + kh * 8]);
#pragma unroll
            for (int ti = 0; ti < 2; ++ti) {
                acc[ti][tj] = __builtin_amdgcn_wmma_f32_16x16x32_bf16(
                    false, ahi[ti], false, bhi, (short)0, acc[ti][tj], false, false);
                acc[ti][tj] = __builtin_amdgcn_wmma_f32_16x16x32_bf16(
                    false, alo[ti], false, bhi, (short)0, acc[ti][tj], false, false);
                acc[ti][tj] = __builtin_amdgcn_wmma_f32_16x16x32_bf16(
                    false, ahi[ti], false, blo, (short)0, acc[ti][tj], false, false);
            }
        }
    }
    __syncthreads();

    // spill h tile (+bias) to LDS for cross-lane blade access
    float (*hs)[132] = (float (*)[132])smem;
#pragma unroll
    for (int ti = 0; ti < 2; ++ti)
#pragma unroll
        for (int tj = 0; tj < 4; ++tj) {
            const int col  = wn * 64 + tj * 16 + l15;
            const float bias = bemb[bN * BN + col];
            const int rbase = wm * 32 + ti * 16 + (kh ? 8 : 0);
#pragma unroll
            for (int v = 0; v < 8; ++v)
                hs[rbase + v][col] = acc[ti][tj][v] + bias;
        }
    __syncthreads();

    clifford_epilogue(hs, red, wgp, Wread, partial, bN, bM, t);
}

// ---------------- fallback: in-kernel conversion (small workspace) ----------------
__global__ __launch_bounds__(256)
void clifford_fused_inline(const float* __restrict__ x, const float* __restrict__ Wemb,
                           const float* __restrict__ bemb, const float* __restrict__ wgp,
                           const float* __restrict__ Wread, float* __restrict__ partial) {
    __shared__ __align__(16) unsigned char smem[BM * 132 * 4];
    __shared__ float red[256][2];
    __bf16 (*sAhi)[40] = (__bf16 (*)[40])(smem);
    __bf16 (*sAlo)[40] = (__bf16 (*)[40])(smem + PLANE_B);
    __bf16 (*sBhi)[40] = (__bf16 (*)[40])(smem + 2 * PLANE_B);
    __bf16 (*sBlo)[40] = (__bf16 (*)[40])(smem + 3 * PLANE_B);

    const int t    = threadIdx.x;
    const int bN   = blockIdx.x;
    const int bM   = blockIdx.y;
    const int lane = t & 31;
    const int wid  = t >> 5;
    const int wm   = wid >> 1;
    const int wn   = wid & 1;
    const int l15  = lane & 15;
    const int kh   = lane >> 4;

    v8f acc[2][4];
#pragma unroll
    for (int ti = 0; ti < 2; ++ti)
#pragma unroll
        for (int tj = 0; tj < 4; ++tj) { v8f z = {}; acc[ti][tj] = z; }

    for (int kk = 0; kk < INFEAT; kk += KS) {
#pragma unroll
        for (int p = 0; p < 4; ++p) {
            const int m  = p * 32 + (t >> 3);
            const int j4 = (t & 7) * 4;
            float4 va = *(const float4*)&x[(size_t)(bM * BM + m) * INFEAT + kk + j4];
            split_store(&sAhi[m][j4], &sAlo[m][j4], va);
            float4 vb = *(const float4*)&Wemb[(size_t)(bN * BN + m) * INFEAT + kk + j4];
            split_store(&sBhi[m][j4], &sBlo[m][j4], vb);
        }
        __syncthreads();
        v16bf ahi[2], alo[2];
#pragma unroll
        for (int ti = 0; ti < 2; ++ti) {
            const int ar = wm * 32 + ti * 16 + l15;
            ahi[ti] = frag16(&sAhi[ar][kh * 8], &sAhi[ar][16 + kh * 8]);
            alo[ti] = frag16(&sAlo[ar][kh * 8], &sAlo[ar][16 + kh * 8]);
        }
#pragma unroll
        for (int tj = 0; tj < 4; ++tj) {
            const int br = wn * 64 + tj * 16 + l15;
            v16bf bhi = frag16(&sBhi[br][kh * 8], &sBhi[br][16 + kh * 8]);
            v16bf blo = frag16(&sBlo[br][kh * 8], &sBlo[br][16 + kh * 8]);
#pragma unroll
            for (int ti = 0; ti < 2; ++ti) {
                acc[ti][tj] = __builtin_amdgcn_wmma_f32_16x16x32_bf16(
                    false, ahi[ti], false, bhi, (short)0, acc[ti][tj], false, false);
                acc[ti][tj] = __builtin_amdgcn_wmma_f32_16x16x32_bf16(
                    false, alo[ti], false, bhi, (short)0, acc[ti][tj], false, false);
                acc[ti][tj] = __builtin_amdgcn_wmma_f32_16x16x32_bf16(
                    false, ahi[ti], false, blo, (short)0, acc[ti][tj], false, false);
            }
        }
        __syncthreads();
    }

    float (*hs)[132] = (float (*)[132])smem;
#pragma unroll
    for (int ti = 0; ti < 2; ++ti)
#pragma unroll
        for (int tj = 0; tj < 4; ++tj) {
            const int col  = wn * 64 + tj * 16 + l15;
            const float bias = bemb[bN * BN + col];
            const int rbase = wm * 32 + ti * 16 + (kh ? 8 : 0);
#pragma unroll
            for (int v = 0; v < 8; ++v)
                hs[rbase + v][col] = acc[ti][tj][v] + bias;
        }
    __syncthreads();

    clifford_epilogue(hs, red, wgp, Wread, partial, bN, bM, t);
}

// ---------------- deterministic final reduction ----------------
__global__ __launch_bounds__(256)
void clifford_reduce(const float* __restrict__ partial,
                     const float* __restrict__ bread,
                     float* __restrict__ out) {
    const int idx = blockIdx.x * 256 + threadIdx.x;   // b*2+r
    float s = bread[idx & 1];
#pragma unroll 8
    for (int nb = 0; nb < NBLK; ++nb)
        s += partial[(size_t)nb * (BATCH * 2) + idx];
    out[idx] = s;
}

extern "C" void kernel_launch(void* const* d_in, const int* in_sizes, int n_in,
                              void* d_out, int out_size, void* d_ws, size_t ws_size,
                              hipStream_t stream) {
    const float* x     = (const float*)d_in[0];
    const float* Wemb  = (const float*)d_in[1];
    const float* bemb  = (const float*)d_in[2];
    const float* wgp   = (const float*)d_in[3];
    const float* Wread = (const float*)d_in[4];
    const float* bread = (const float*)d_in[5];
    float* out     = (float*)d_out;
    float* partial = (float*)d_ws;

    const size_t PARTIAL_BYTES = (size_t)NBLK * BATCH * 2 * sizeof(float);  // 4 MB
    const size_t PLANE_ELEMS   = (size_t)BATCH * INFEAT;                    // == NTOT*INFEAT
    const size_t PLANE_BYTES   = PLANE_ELEMS * sizeof(__bf16);              // 16 MB
    const size_t FULL_BYTES    = PARTIAL_BYTES + 4 * PLANE_BYTES;           // ~71 MB

    dim3 grid(NBLK, MBLK);

    if (ws_size >= FULL_BYTES) {
        __bf16* Ahi = (__bf16*)((char*)d_ws + PARTIAL_BYTES);
        __bf16* Alo = Ahi + PLANE_ELEMS;
        __bf16* Bhi = Alo + PLANE_ELEMS;
        __bf16* Blo = Bhi + PLANE_ELEMS;
        const int splitBlocks = (int)(PLANE_ELEMS / (256 * 4));   // 8192
        bf16_split<<<splitBlocks, 256, 0, stream>>>(x,    Ahi, Alo);
        bf16_split<<<splitBlocks, 256, 0, stream>>>(Wemb, Bhi, Blo);
        clifford_fused_async<<<grid, 256, 0, stream>>>(Ahi, Alo, Bhi, Blo,
                                                       bemb, wgp, Wread, partial);
    } else if (ws_size >= PARTIAL_BYTES) {
        clifford_fused_inline<<<grid, 256, 0, stream>>>(x, Wemb, bemb, wgp,
                                                        Wread, partial);
    } else {
        return;
    }
    clifford_reduce<<<BATCH * 2 / 256, 256, 0, stream>>>(partial, bread, out);
}
